// NaiveMoE_75299366633544
// MI455X (gfx1250) — matrix operand order repaired and checked by reading, not compiled, not run on hardware
//
#include <hip/hip_runtime.h>

// ---------------- problem constants ----------------
#define TT 8192      // tokens = B*S
#define DD 512       // model dim
#define FF 2048      // ffn dim
#define EE 8         // experts
#define TM 128       // tile M
#define TN 128       // tile N
#define TK 32        // tile K (bf16 wmma depth)

typedef __attribute__((ext_vector_type(16))) __bf16 bf16x16;
typedef __attribute__((ext_vector_type(8)))  float  floatx8;

union Frag { bf16x16 v; uint4 q[2]; };

__device__ __forceinline__ unsigned short f32_to_bf16(float f) {
    unsigned int u = __float_as_uint(f);
    u += 0x7FFFu + ((u >> 16) & 1u);   // round-to-nearest-even
    return (unsigned short)(u >> 16);
}

__device__ __forceinline__ unsigned int pack2_bf16(float a, float b) {
    return (unsigned)f32_to_bf16(a) | ((unsigned)f32_to_bf16(b) << 16);
}

__device__ __forceinline__ float gelu_exact(float v) {
    return 0.5f * v * (1.0f + erff(v * 0.70710678118654752440f));
}

// A fragment: 16x32 bf16, lane (m = lane&15, h = lane>>4) holds
// K = {8h..8h+7} then {16+8h..23+8h}  (ISA 7.12.2, 16-bit A 16x32)
__device__ __forceinline__ bf16x16 load_a_frag(const unsigned short* As, int mbase, int lane) {
    int m = mbase + (lane & 15);
    int h = lane >> 4;
    const unsigned short* p = As + m * TK;
    Frag f;
    f.q[0] = *(const uint4*)(p + 8 * h);
    f.q[1] = *(const uint4*)(p + 16 + 8 * h);
    return f.v;
}

// B fragment: 32x16 bf16 stored N-major in LDS (Bs[n][k]); lane (n, h) holds
// K = {8h..8h+7} then {16+8h..23+8h}
__device__ __forceinline__ bf16x16 load_b_frag(const unsigned short* Bs, int nbase, int lane) {
    int n = nbase + (lane & 15);
    int h = lane >> 4;
    const unsigned short* p = Bs + n * TK;
    Frag f;
    f.q[0] = *(const uint4*)(p + 8 * h);
    f.q[1] = *(const uint4*)(p + 16 + 8 * h);
    return f.v;
}

// ---------------- prep: zero counts ----------------
__global__ void moe_zero_kernel(int* __restrict__ cnt) {
    if (threadIdx.x < EE) cnt[threadIdx.x] = 0;
}

// ---------------- prep: x (f32) -> xb (bf16), 8 elems/thread ----------------
__global__ void moe_cvt_x_kernel(const float* __restrict__ x, unsigned short* __restrict__ xb) {
    size_t base = ((size_t)blockIdx.x * 256 + threadIdx.x) * 8;
    float4 a = *(const float4*)(x + base);
    float4 b = *(const float4*)(x + base + 4);
    uint4 r;
    r.x = pack2_bf16(a.x, a.y);
    r.y = pack2_bf16(a.z, a.w);
    r.z = pack2_bf16(b.x, b.y);
    r.w = pack2_bf16(b.z, b.w);
    *(uint4*)(xb + base) = r;
}

// ---------------- prep: per-expert transpose f32 [R][C] -> bf16 [C][R] ----------------
__global__ void moe_transpose_kernel(const float* __restrict__ src, unsigned short* __restrict__ dst,
                                     int R, int C) {
    __shared__ unsigned short tile[32][33];
    const float* s = src + (size_t)blockIdx.z * R * C;
    unsigned short* d = dst + (size_t)blockIdx.z * R * C;
    int c0 = blockIdx.x * 32, r0 = blockIdx.y * 32;
    int tx = threadIdx.x, ty = threadIdx.y;
#pragma unroll
    for (int i = ty; i < 32; i += 8)
        tile[i][tx] = f32_to_bf16(s[(size_t)(r0 + i) * C + c0 + tx]);
    __syncthreads();
#pragma unroll
    for (int i = ty; i < 32; i += 8)
        d[(size_t)(c0 + i) * R + r0 + tx] = tile[tx][i];
}

// ---------------- router (one wave32 per token) ----------------
__global__ void moe_router_kernel(const float* __restrict__ x, const float* __restrict__ gw,
                                  float* __restrict__ probs, int* __restrict__ cnt,
                                  int* __restrict__ tokIdx, int* __restrict__ selEP,
                                  float* __restrict__ selW) {
    int wave = threadIdx.x >> 5;
    int lane = threadIdx.x & 31;
    int t = blockIdx.x * 8 + wave;
    if (t >= TT) return;

    float acc[EE];
#pragma unroll
    for (int e = 0; e < EE; ++e) acc[e] = 0.0f;

    const float* xrow = x + (size_t)t * DD + lane * 16;
#pragma unroll 4
    for (int i = 0; i < 16; ++i) {
        float xv = xrow[i];
        const float* g = gw + (size_t)(lane * 16 + i) * EE;
#pragma unroll
        for (int e = 0; e < EE; ++e) acc[e] = fmaf(xv, g[e], acc[e]);
    }
#pragma unroll
    for (int off = 16; off >= 1; off >>= 1) {
#pragma unroll
        for (int e = 0; e < EE; ++e) acc[e] += __shfl_xor(acc[e], off);
    }
    if (lane == 0) {
        float mx = acc[0];
#pragma unroll
        for (int e = 1; e < EE; ++e) mx = fmaxf(mx, acc[e]);
        float p[EE]; float s = 0.0f;
#pragma unroll
        for (int e = 0; e < EE; ++e) { p[e] = expf(acc[e] - mx); s += p[e]; }
        float inv = 1.0f / s;
#pragma unroll
        for (int e = 0; e < EE; ++e) p[e] *= inv;
        *(float4*)(probs + (size_t)t * EE)     = make_float4(p[0], p[1], p[2], p[3]);
        *(float4*)(probs + (size_t)t * EE + 4) = make_float4(p[4], p[5], p[6], p[7]);
        // top-2 (ties -> lower index, matching lax.top_k)
        int i0 = 0;
#pragma unroll
        for (int e = 1; e < EE; ++e) if (p[e] > p[i0]) i0 = e;
        int i1 = (i0 == 0) ? 1 : 0;
#pragma unroll
        for (int e = 0; e < EE; ++e) if (e != i0 && p[e] > p[i1]) i1 = e;
        float wsum = p[i0] + p[i1];
        int s0 = atomicAdd(&cnt[i0], 1);
        tokIdx[i0 * TT + s0] = t;
        int s1 = atomicAdd(&cnt[i1], 1);
        tokIdx[i1 * TT + s1] = t;
        selEP[t * 2 + 0] = i0 * TT + s0;   // pack expert*TT + pos
        selEP[t * 2 + 1] = i1 * TT + s1;
        selW[t * 2 + 0] = p[i0] / wsum;
        selW[t * 2 + 1] = p[i1] / wsum;
    }
}

// ---------------- deterministic loss + offset scan ----------------
__global__ void moe_scan_loss_kernel(const float* __restrict__ probs, const int* __restrict__ cnt,
                                     int* __restrict__ off, float* __restrict__ loss_out) {
    __shared__ float sh[EE * 256];
    int tid = threadIdx.x;
    float a[EE];
#pragma unroll
    for (int e = 0; e < EE; ++e) a[e] = 0.0f;
    for (int t = tid; t < TT; t += 256) {
        const float* pr = probs + (size_t)t * EE;
#pragma unroll
        for (int e = 0; e < EE; ++e) a[e] += pr[e];
    }
#pragma unroll
    for (int e = 0; e < EE; ++e) sh[e * 256 + tid] = a[e];
    __syncthreads();
    for (int s = 128; s > 0; s >>= 1) {
        if (tid < s) {
#pragma unroll
            for (int e = 0; e < EE; ++e) sh[e * 256 + tid] += sh[e * 256 + tid + s];
        }
        __syncthreads();
    }
    if (tid == 0) {
        float lb = 0.0f;
#pragma unroll
        for (int e = 0; e < EE; ++e) {
            float m = sh[e * 256] / (float)TT;
            lb += m * m;
        }
        *loss_out = (float)EE * lb;
        int o = 0;
        for (int e = 0; e < EE; ++e) { off[e] = o; o += cnt[e]; }
        off[EE] = o;
    }
}

// ---------------- GEMM1: H = gelu(Xb_gather @ W1T_e^T + b1_e), bf16 out ----------------
__global__ void __launch_bounds__(256)
moe_ffn1_kernel(const unsigned short* __restrict__ xb, const unsigned short* __restrict__ w1T,
                const float* __restrict__ b1, const int* __restrict__ cnt,
                const int* __restrict__ off, const int* __restrict__ tokIdx,
                unsigned short* __restrict__ H) {
    __shared__ unsigned short As[TM * TK];
    __shared__ unsigned short Bs[TN * TK];
    __shared__ int toks[TM];

    const int e  = blockIdx.x >> 6;       // 64 M-tiles per expert
    const int mt = blockIdx.x & 63;
    const int c  = cnt[e];
    if (mt * TM >= c) return;
    const int f0 = blockIdx.y * TN;
    const int hb = off[e];

    const int tid  = threadIdx.x;
    const int lane = tid & 31;
    const int wave = tid >> 5;
    const int wm = wave & 1;              // 2 x 64 rows
    const int wn = wave >> 1;             // 4 x 32 cols

    if (tid < TM) toks[tid] = tokIdx[e * TT + min(mt * TM + tid, c - 1)];
    __syncthreads();

    // per-thread fill: 2 octs (8 bf16) for A, 2 for B; all uint4 copies
    const int kO = (tid & 3) * 8;
    int mO[2];
    const unsigned short* aptr[2];
    const unsigned short* bptr[2];
#pragma unroll
    for (int p = 0; p < 2; ++p) {
        mO[p] = (tid >> 2) + p * 64;
        aptr[p] = xb + (size_t)toks[mO[p]] * DD + kO;
        bptr[p] = w1T + ((size_t)e * FF + f0 + mO[p]) * DD + kO;   // w1T: [E][F][D]
    }

    floatx8 zero8 = {0.f,0.f,0.f,0.f,0.f,0.f,0.f,0.f};
    floatx8 acc[4][2];
#pragma unroll
    for (int i = 0; i < 4; ++i)
#pragma unroll
        for (int j = 0; j < 2; ++j) acc[i][j] = zero8;

    uint4 ra[2], rb[2];
#pragma unroll
    for (int p = 0; p < 2; ++p) { ra[p] = *(const uint4*)(aptr[p]); rb[p] = *(const uint4*)(bptr[p]); }

    const int NS = DD / TK;               // 16 K-steps
    for (int s = 0; s < NS; ++s) {
        __syncthreads();                  // previous compute done reading LDS
#pragma unroll
        for (int p = 0; p < 2; ++p) {
            *(uint4*)&As[mO[p] * TK + kO] = ra[p];
            *(uint4*)&Bs[mO[p] * TK + kO] = rb[p];
        }
        __syncthreads();                  // LDS tile visible
        if (s + 1 < NS) {                 // prefetch next K-tile (overlaps WMMA)
            int kk = (s + 1) * TK;
#pragma unroll
            for (int p = 0; p < 2; ++p) {
                ra[p] = *(const uint4*)(aptr[p] + kk);
                rb[p] = *(const uint4*)(bptr[p] + kk);
            }
        }
        bf16x16 af[4];
#pragma unroll
        for (int i = 0; i < 4; ++i) af[i] = load_a_frag(As, wm * 64 + i * 16, lane);
#pragma unroll
        for (int j = 0; j < 2; ++j) {
            bf16x16 bf = load_b_frag(Bs, wn * 32 + j * 16, lane);
#pragma unroll
            for (int i = 0; i < 4; ++i)
                acc[i][j] = __builtin_amdgcn_wmma_f32_16x16x32_bf16(
                    false, af[i], false, bf, (short)0, acc[i][j], false, false);
        }
    }

    const float* b1e = b1 + (size_t)e * FF;
    const int h   = lane >> 4;
    const int col = lane & 15;
#pragma unroll
    for (int i = 0; i < 4; ++i) {
#pragma unroll
        for (int j = 0; j < 2; ++j) {
#pragma unroll
            for (int r = 0; r < 8; ++r) {
                int m_in = wm * 64 + i * 16 + r + 8 * h;
                int row  = mt * TM + m_in;
                if (row < c) {
                    int f = f0 + wn * 32 + j * 16 + col;
                    float v = acc[i][j][r] + b1e[f];
                    H[(size_t)(hb + row) * FF + f] = f32_to_bf16(gelu_exact(v));
                }
            }
        }
    }
}

// ---------------- GEMM2: outS = H @ W2T_e^T + b2_e (per slot, plain stores) ----------------
__global__ void __launch_bounds__(256)
moe_ffn2_kernel(const unsigned short* __restrict__ H, const unsigned short* __restrict__ w2T,
                const float* __restrict__ b2, const int* __restrict__ cnt,
                const int* __restrict__ off, float* __restrict__ outS) {
    __shared__ unsigned short As[TM * TK];
    __shared__ unsigned short Bs[TN * TK];

    const int e  = blockIdx.x >> 6;
    const int mt = blockIdx.x & 63;
    const int c  = cnt[e];
    if (mt * TM >= c) return;
    const int d0 = blockIdx.y * TN;       // D/128 = 4 N-tiles
    const int hb = off[e];

    const int tid  = threadIdx.x;
    const int lane = tid & 31;
    const int wave = tid >> 5;
    const int wm = wave & 1;
    const int wn = wave >> 1;

    const int kO = (tid & 3) * 8;
    int mO[2];
    const unsigned short* aptr[2];
    const unsigned short* bptr[2];
#pragma unroll
    for (int p = 0; p < 2; ++p) {
        mO[p] = (tid >> 2) + p * 64;
        int row = min(mt * TM + mO[p], c - 1);
        aptr[p] = H + (size_t)(hb + row) * FF + kO;
        bptr[p] = w2T + ((size_t)e * DD + d0 + mO[p]) * FF + kO;   // w2T: [E][D][F]
    }

    floatx8 zero8 = {0.f,0.f,0.f,0.f,0.f,0.f,0.f,0.f};
    floatx8 acc[4][2];
#pragma unroll
    for (int i = 0; i < 4; ++i)
#pragma unroll
        for (int j = 0; j < 2; ++j) acc[i][j] = zero8;

    uint4 ra[2], rb[2];
#pragma unroll
    for (int p = 0; p < 2; ++p) { ra[p] = *(const uint4*)(aptr[p]); rb[p] = *(const uint4*)(bptr[p]); }

    const int NS = FF / TK;               // 64 K-steps
    for (int s = 0; s < NS; ++s) {
        __syncthreads();
#pragma unroll
        for (int p = 0; p < 2; ++p) {
            *(uint4*)&As[mO[p] * TK + kO] = ra[p];
            *(uint4*)&Bs[mO[p] * TK + kO] = rb[p];
        }
        __syncthreads();
        if (s + 1 < NS) {
            int kk = (s + 1) * TK;
#pragma unroll
            for (int p = 0; p < 2; ++p) {
                ra[p] = *(const uint4*)(aptr[p] + kk);
                rb[p] = *(const uint4*)(bptr[p] + kk);
            }
        }
        bf16x16 af[4];
#pragma unroll
        for (int i = 0; i < 4; ++i) af[i] = load_a_frag(As, wm * 64 + i * 16, lane);
#pragma unroll
        for (int j = 0; j < 2; ++j) {
            bf16x16 bf = load_b_frag(Bs, wn * 32 + j * 16, lane);
#pragma unroll
            for (int i = 0; i < 4; ++i)
                acc[i][j] = __builtin_amdgcn_wmma_f32_16x16x32_bf16(
                    false, af[i], false, bf, (short)0, acc[i][j], false, false);
        }
    }

    const float* b2e = b2 + (size_t)e * DD;
    const int h   = lane >> 4;
    const int col = lane & 15;
#pragma unroll
    for (int i = 0; i < 4; ++i) {
#pragma unroll
        for (int j = 0; j < 2; ++j) {
#pragma unroll
            for (int r = 0; r < 8; ++r) {
                int m_in = wm * 64 + i * 16 + r + 8 * h;
                int row  = mt * TM + m_in;
                if (row < c) {
                    int d = d0 + wn * 32 + j * 16 + col;
                    outS[(size_t)(hb + row) * DD + d] = acc[i][j][r] + b2e[d];
                }
            }
        }
    }
}

// ---------------- deterministic combine y[t] = w0*outS[s0] + w1*outS[s1] ----------------
__global__ void moe_combine_kernel(const float* __restrict__ outS, const int* __restrict__ selEP,
                                   const float* __restrict__ selW, const int* __restrict__ off,
                                   float* __restrict__ y) {
    int gid = blockIdx.x * 256 + threadIdx.x;   // TT*DD/4 quads
    int t  = gid >> 7;                          // 128 quads per token
    int dq = (gid & 127) << 2;
    int sp0 = selEP[t * 2], sp1 = selEP[t * 2 + 1];
    float w0 = selW[t * 2], w1 = selW[t * 2 + 1];
    int e0 = sp0 >> 13, p0 = sp0 & (TT - 1);
    int e1 = sp1 >> 13, p1 = sp1 & (TT - 1);
    const float4 a = *(const float4*)(outS + (size_t)(off[e0] + p0) * DD + dq);
    const float4 b = *(const float4*)(outS + (size_t)(off[e1] + p1) * DD + dq);
    float4 o;
    o.x = w0 * a.x + w1 * b.x;
    o.y = w0 * a.y + w1 * b.y;
    o.z = w0 * a.z + w1 * b.z;
    o.w = w0 * a.w + w1 * b.w;
    *(float4*)(y + (size_t)t * DD + dq) = o;
}

// ---------------- host launcher ----------------
extern "C" void kernel_launch(void* const* d_in, const int* in_sizes, int n_in,
                              void* d_out, int out_size, void* d_ws, size_t ws_size,
                              hipStream_t stream) {
    (void)in_sizes; (void)n_in; (void)out_size; (void)ws_size;
    const float* x      = (const float*)d_in[0];
    const float* gate_w = (const float*)d_in[1];
    const float* w1     = (const float*)d_in[2];
    const float* b1     = (const float*)d_in[3];
    const float* w2     = (const float*)d_in[4];
    const float* b2     = (const float*)d_in[5];
    float* y = (float*)d_out;

    const size_t MB = 1024ull * 1024ull;
    char* ws = (char*)d_ws;
    int*   cnt    = (int*)(ws + 0);                              // 8 i32
    int*   off    = (int*)(ws + 64);                             // 9 i32
    float* probs  = (float*)(ws + 1024);                         // T*E f32 (256 KB)
    int*   tokIdx = (int*)(ws + 1024 + (size_t)TT * EE * 4);     // E*T i32 (256 KB)
    int*   selEP  = (int*)(ws + 1024 + (size_t)2 * TT * EE * 4); // 2T i32 (64 KB)
    float* selW   = (float*)(ws + 1024 + (size_t)2 * TT * EE * 4 + (size_t)2 * TT * 4);
    unsigned short* xb  = (unsigned short*)(ws + 1 * MB);        // T*D bf16   (8 MB)
    unsigned short* w1T = (unsigned short*)(ws + 9 * MB);        // E*F*D bf16 (16 MB)
    unsigned short* w2T = (unsigned short*)(ws + 25 * MB);       // E*D*F bf16 (16 MB)
    unsigned short* H   = (unsigned short*)(ws + 41 * MB);       // 2T*F bf16  (64 MB)
    float*          outS = (float*)(ws + 105 * MB);              // 2T*D f32   (32 MB)

    moe_zero_kernel<<<dim3(1), 32, 0, stream>>>(cnt);
    moe_cvt_x_kernel<<<dim3(TT * DD / 8 / 256), 256, 0, stream>>>(x, xb);
    moe_transpose_kernel<<<dim3(FF / 32, DD / 32, EE), dim3(32, 8), 0, stream>>>(w1, w1T, DD, FF);
    moe_transpose_kernel<<<dim3(DD / 32, FF / 32, EE), dim3(32, 8), 0, stream>>>(w2, w2T, FF, DD);
    moe_router_kernel<<<dim3(TT / 8), 256, 0, stream>>>(x, gate_w, probs, cnt, tokIdx, selEP, selW);
    moe_scan_loss_kernel<<<dim3(1), 256, 0, stream>>>(probs, cnt, off, y + (size_t)TT * DD);
    moe_ffn1_kernel<<<dim3(EE * (TT / TM), FF / TN), 256, 0, stream>>>(xb, w1T, b1, cnt, off, tokIdx, H);
    moe_ffn2_kernel<<<dim3(EE * (TT / TM), DD / TN), 256, 0, stream>>>(H, w2T, b2, cnt, off, outS);
    moe_combine_kernel<<<dim3((TT * DD / 4) / 256), 256, 0, stream>>>(outS, selEP, selW, off, y);
}
